// EMGCombinedModel_72241349919398
// MI455X (gfx1250) — matrix operand order repaired and checked by reading, not compiled, not run on hardware
//
#include <hip/hip_runtime.h>
#include <hip/hip_bf16.h>

typedef __attribute__((ext_vector_type(16))) _Float16 v16h;
typedef __attribute__((ext_vector_type(8)))  _Float16 v8h;
typedef __attribute__((ext_vector_type(8)))  float    v8f;

#define BATCH  2048
#define TSTEPS 512
#define HID    64
#define GATES  256   // 4*HID

// LDS budget (bytes):
//  3 weight mats f16 [256][64]      : 3*32768 = 98304
//  gb0,gb1,wih0c f32 [256]          : 3*1024  = 3072     -> 101376
//  fc1w f32 [64*66]                 : 16896              -> 118272
//  fc1b[64], fc2w[64], fc2b+pad[16] : 576                -> 118848
//  h1buf[2][16][64] f16             : 4096               -> 122944
//  h2buf[2][16][64] f16             : 4096               -> 127040
//  xt[2][16] f32                    : 128                -> 127168
#define SMEM_BYTES 127168

__device__ __forceinline__ float sigf(float x) { return 1.0f / (1.0f + __expf(-x)); }
__device__ __forceinline__ float tanh_fast(float x) {
  x = fminf(fmaxf(x, -15.f), 15.f);
  float e = __expf(2.f * x);
  return (e - 1.f) / (e + 1.f);
}

__device__ __forceinline__ v8f wmma_f16(v16h a, v16h b, v8f c) {
  // v_wmma_f32_16x16x32_f16: (neg_a, A, neg_b, B, c_mod, C, reuse_a, reuse_b)
  return __builtin_amdgcn_wmma_f32_16x16x32_f16(false, a, false, b, (short)0, c, false, false);
}

// A fragment: 16x32 f16, hbuf is [16 rows M][64 cols K] in LDS.
// Lanes 0-15: K = {kb+0..7, kb+16..23}; lanes 16-31: K = {kb+8..15, kb+24..31}.
__device__ __forceinline__ v16h ldA(const _Float16* hbuf, int kb, int lh, int hi) {
  const v8h lo = *(const v8h*)(hbuf + lh * HID + kb + hi * 8);
  const v8h hh = *(const v8h*)(hbuf + lh * HID + kb + 16 + hi * 8);
  return __builtin_shufflevector(lo, hh, 0,1,2,3,4,5,6,7,8,9,10,11,12,13,14,15);
}

// B fragment: 32x16 f16 of W^T (W stored [g][k], g = gate column = N, k = K).
// Lane n holds column N = ncol+lh, K = {kb+hi*8+0..7, kb+16+hi*8+0..7}.
__device__ __forceinline__ v16h ldB(const _Float16* w, int ncol, int kb, int lh, int hi) {
  const v8h lo = *(const v8h*)(w + (ncol + lh) * HID + kb + hi * 8);
  const v8h hh = *(const v8h*)(w + (ncol + lh) * HID + kb + 16 + hi * 8);
  return __builtin_shufflevector(lo, hh, 0,1,2,3,4,5,6,7,8,9,10,11,12,13,14,15);
}

__global__ void __launch_bounds__(128)
lstm2_fused_wmma_kernel(const float* __restrict__ x,      // [B,T]
                        const float* __restrict__ W_ih0,  // [256,1]
                        const float* __restrict__ W_hh0,  // [256,64]
                        const float* __restrict__ b_ih0,
                        const float* __restrict__ b_hh0,
                        const float* __restrict__ W_ih1,  // [256,64]
                        const float* __restrict__ W_hh1,  // [256,64]
                        const float* __restrict__ b_ih1,
                        const float* __restrict__ b_hh1,
                        const float* __restrict__ fc1_w,  // [64,66]
                        const float* __restrict__ fc1_b,  // [64]
                        const float* __restrict__ fc2_w,  // [64]
                        const float* __restrict__ fc2_b,  // [1]
                        float* __restrict__ out)          // [B]
{
  extern __shared__ __align__(128) char smem[];
  _Float16* whh0  = (_Float16*)(smem);
  _Float16* wih1  = whh0 + GATES * HID;
  _Float16* whh1  = wih1 + GATES * HID;
  float*    gb0   = (float*)(whh1 + GATES * HID);
  float*    gb1   = gb0 + GATES;
  float*    wih0c = gb1 + GATES;
  float*    fc1w  = wih0c + GATES;
  float*    fc1b  = fc1w + 64 * 66;
  float*    fc2w  = fc1b + 64;
  float*    fc2bp = fc2w + 64;                 // 1 used + pad to 16
  _Float16* h1buf = (_Float16*)(fc2bp + 16);   // [2][16][64] ping-pong
  _Float16* h2buf = h1buf + 2 * 16 * HID;      // [2][16][64] ping-pong
  float*    xt    = (float*)(h2buf + 2 * 16 * HID); // [2][16] ping-pong

  const int tid  = threadIdx.x;
  const int lane = tid & 31;
  const int wave = tid >> 5;       // 0..3 : gate-column slice owner (one per SIMD32)
  const int lh   = lane & 15;
  const int hi   = lane >> 4;
  const int rb   = blockIdx.x * 16;

  // ---- one-time weight load + f32->f16 conversion into LDS (all 4 waves) ----
  for (int i = tid; i < GATES * HID; i += 128) {
    whh0[i] = (_Float16)W_hh0[i];
    wih1[i] = (_Float16)W_ih1[i];
    whh1[i] = (_Float16)W_hh1[i];
  }
  for (int g = tid; g < GATES; g += 128) {
    gb0[g]   = b_ih0[g] + b_hh0[g];
    gb1[g]   = b_ih1[g] + b_hh1[g];
    wih0c[g] = W_ih0[g];
  }
  for (int i = tid; i < 64 * 66; i += 128) fc1w[i] = fc1_w[i];
  if (tid < 64) { fc1b[tid] = fc1_b[tid]; fc2w[tid] = fc2_w[tid]; }
  if (tid == 0) fc2bp[0] = fc2_b[0];
  for (int i = tid; i < 2 * 16 * HID; i += 128) {
    h1buf[i] = (_Float16)0.f; h2buf[i] = (_Float16)0.f;
  }
  __syncthreads();

  // ---- hoist this wave's loop-invariant B fragments into VGPRs ----
  // Wave w owns columns [w*16, w*16+16) of each of the 4 gates, both layers.
  v16h Bhh0[4][2], Bih1[4][2], Bhh1[4][2];   // [gate i/f/g/o][K-chunk]
  #pragma unroll
  for (int g = 0; g < 4; ++g) {
    #pragma unroll
    for (int k = 0; k < 2; ++k) {
      Bhh0[g][k] = ldB(whh0, g * 64 + wave * 16, k * 32, lh, hi);
      Bih1[g][k] = ldB(wih1, g * 64 + wave * 16, k * 32, lh, hi);
      Bhh1[g][k] = ldB(whh1, g * 64 + wave * 16, k * 32, lh, hi);
    }
  }
  // loop-invariant per-lane bias / input-weight scalars
  const int nc = wave * 16 + lh;                 // column within each gate block
  const float wi0 = wih0c[nc],       bi0 = gb0[nc];
  const float wf0 = wih0c[64 + nc],  bf0 = gb0[64 + nc];
  const float wg0 = wih0c[128 + nc], bg0 = gb0[128 + nc];
  const float wo0 = wih0c[192 + nc], bo0 = gb0[192 + nc];
  const float bi1 = gb1[nc], bf1 = gb1[64 + nc], bg1 = gb1[128 + nc], bo1 = gb1[192 + nc];

  float c1[8], c2[8];
  #pragma unroll
  for (int r = 0; r < 8; ++r) { c1[r] = 0.f; c2[r] = 0.f; }

  float sumsq = 0.f, sumabs = 0.f;
  const float* xrow = x + (size_t)(rb + lh) * TSTEPS;

  int cur = 0;
  for (int t = 0; t < TSTEPS; ++t, cur ^= 1) {
    const int nxt = cur ^ 1;
    if (wave == 0) {
      float xv = xrow[t];
      __builtin_prefetch(xrow + t + 32, 0, 0);   // global_prefetch_b8 ahead of scan
      sumsq  += xv * xv;
      sumabs += fabsf(xv);
      xt[cur * 16 + lh] = xv;
    }
    __syncthreads();   // B_top: xt[cur] ready; h1buf/h2buf[cur] from prev step visible

    float xr[8];
    #pragma unroll
    for (int r = 0; r < 8; ++r) xr[r] = xt[cur * 16 + hi * 8 + r];

    const _Float16* h1c = h1buf + cur * 16 * HID;
    const _Float16* h2c = h2buf + cur * 16 * HID;
    v16h h1a0 = ldA(h1c, 0, lh, hi);
    v16h h1a1 = ldA(h1c, 32, lh, hi);
    v16h h2a0 = ldA(h2c, 0, lh, hi);
    v16h h2a1 = ldA(h2c, 32, lh, hi);

    // ================= layer 0 (this wave's 16-column slice of each gate) ====
    v8f gi = {}, gf = {}, gg = {}, go = {};
    gi = wmma_f16(h1a0, Bhh0[0][0], gi); gi = wmma_f16(h1a1, Bhh0[0][1], gi);
    gf = wmma_f16(h1a0, Bhh0[1][0], gf); gf = wmma_f16(h1a1, Bhh0[1][1], gf);
    gg = wmma_f16(h1a0, Bhh0[2][0], gg); gg = wmma_f16(h1a1, Bhh0[2][1], gg);
    go = wmma_f16(h1a0, Bhh0[3][0], go); go = wmma_f16(h1a1, Bhh0[3][1], go);

    _Float16* h1n = h1buf + nxt * 16 * HID;
    #pragma unroll
    for (int r = 0; r < 8; ++r) {
      const float xm = xr[r];
      const float vi = gi[r] + xm * wi0 + bi0;
      const float vf = gf[r] + xm * wf0 + bf0;
      const float vg = gg[r] + xm * wg0 + bg0;
      const float vo = go[r] + xm * wo0 + bo0;
      const float cn = sigf(vf) * c1[r] + sigf(vi) * tanh_fast(vg);
      c1[r] = cn;
      const float hn = sigf(vo) * tanh_fast(cn);
      h1n[(r + hi * 8) * HID + wave * 16 + lh] = (_Float16)hn;  // C-layout -> [M][K]
    }
    __syncthreads();   // B_mid: full new h1 assembled by all 4 waves

    v16h h1n0 = ldA(h1n, 0, lh, hi);
    v16h h1n1 = ldA(h1n, 32, lh, hi);

    // ================= layer 1 ===============================================
    v8f qi = {}, qf = {}, qg = {}, qo = {};
    qi = wmma_f16(h1n0, Bih1[0][0], qi); qi = wmma_f16(h1n1, Bih1[0][1], qi);
    qi = wmma_f16(h2a0, Bhh1[0][0], qi); qi = wmma_f16(h2a1, Bhh1[0][1], qi);
    qf = wmma_f16(h1n0, Bih1[1][0], qf); qf = wmma_f16(h1n1, Bih1[1][1], qf);
    qf = wmma_f16(h2a0, Bhh1[1][0], qf); qf = wmma_f16(h2a1, Bhh1[1][1], qf);
    qg = wmma_f16(h1n0, Bih1[2][0], qg); qg = wmma_f16(h1n1, Bih1[2][1], qg);
    qg = wmma_f16(h2a0, Bhh1[2][0], qg); qg = wmma_f16(h2a1, Bhh1[2][1], qg);
    qo = wmma_f16(h1n0, Bih1[3][0], qo); qo = wmma_f16(h1n1, Bih1[3][1], qo);
    qo = wmma_f16(h2a0, Bhh1[3][0], qo); qo = wmma_f16(h2a1, Bhh1[3][1], qo);

    _Float16* h2n = h2buf + nxt * 16 * HID;
    #pragma unroll
    for (int r = 0; r < 8; ++r) {
      const float vi = qi[r] + bi1;
      const float vf = qf[r] + bf1;
      const float vg = qg[r] + bg1;
      const float vo = qo[r] + bo1;
      const float cn = sigf(vf) * c2[r] + sigf(vi) * tanh_fast(vg);
      c2[r] = cn;
      const float hn = sigf(vo) * tanh_fast(cn);
      h2n[(r + hi * 8) * HID + wave * 16 + lh] = (_Float16)hn;
    }
    // loop-top __syncthreads makes h2buf[nxt] (== next cur) visible
  }
  __syncthreads();

  // ---- MLP head: lanes 0..15 of wave 0 (they also own the streaming feats) ----
  if (tid < 16) {
    const int m = tid;
    const float rms = sqrtf(sumsq / (float)TSTEPS);
    const float mav = sumabs / (float)TSTEPS;
    const _Float16* h2fin = h2buf + cur * 16 * HID;   // buffer written by last step
    float acc = fc2bp[0];
    for (int k = 0; k < 64; ++k) {
      float s = fc1b[k];
      const float* wr = fc1w + k * 66;
      for (int jj = 0; jj < HID; ++jj)
        s += wr[jj] * (float)h2fin[m * HID + jj];
      s += wr[64] * rms + wr[65] * mav;
      acc += fc2w[k] * fmaxf(s, 0.f);
    }
    out[rb + m] = acc;
  }
}

extern "C" void kernel_launch(void* const* d_in, const int* in_sizes, int n_in,
                              void* d_out, int out_size, void* d_ws, size_t ws_size,
                              hipStream_t stream) {
  const float* x    = (const float*)d_in[0];
  const float* Wih0 = (const float*)d_in[1];
  const float* Whh0 = (const float*)d_in[2];
  const float* bih0 = (const float*)d_in[3];
  const float* bhh0 = (const float*)d_in[4];
  const float* Wih1 = (const float*)d_in[5];
  const float* Whh1 = (const float*)d_in[6];
  const float* bih1 = (const float*)d_in[7];
  const float* bhh1 = (const float*)d_in[8];
  const float* f1w  = (const float*)d_in[9];
  const float* f1b  = (const float*)d_in[10];
  const float* f2w  = (const float*)d_in[11];
  const float* f2b  = (const float*)d_in[12];
  float* out = (float*)d_out;

  dim3 grid(BATCH / 16), block(128);
  hipLaunchKernelGGL(lstm2_fused_wmma_kernel, grid, block, SMEM_BYTES, stream,
                     x, Wih0, Whh0, bih0, bhh0, Wih1, Whh1, bih1, bhh1,
                     f1w, f1b, f2w, f2b, out);
}
